// CPSFMemcellFusedReal_89558658056666
// MI455X (gfx1250) — matrix operand order repaired and checked by reading, not compiled, not run on hardware
//
#include <hip/hip_runtime.h>
#include <math.h>

#define B_DIM 512
#define M_DIM 4096
#define N_DIM 64
#define S_DIM 256

#define BK 32    // K tile per step
#define BKP 40   // padded K stride in LDS (bf16 elems) to spread banks

typedef __attribute__((ext_vector_type(16))) __bf16 v16bf;
typedef __attribute__((ext_vector_type(8)))  __bf16 v8bf;
typedef __attribute__((ext_vector_type(8)))  float  v8f;
typedef int v4i_g __attribute__((vector_size(16)));  // matches builtin pointee

#if defined(__has_builtin)
#if __has_builtin(__builtin_amdgcn_global_load_async_to_lds_b128) && \
    __has_builtin(__builtin_amdgcn_s_wait_asynccnt)
#define HAVE_ASYNC_LDS 1
#endif
#endif
#ifndef HAVE_ASYNC_LDS
#define HAVE_ASYNC_LDS 0
#endif

// One 16-byte global->LDS staging copy. Async DMA path on gfx1250 toolchains
// that expose the builtin (tracked by ASYNCcnt), sync fallback otherwise.
static __device__ inline void stage_b128(const __bf16* g, __bf16* l) {
#if HAVE_ASYNC_LDS
  typedef __attribute__((address_space(1))) v4i_g* gp_t;
  typedef __attribute__((address_space(3))) v4i_g* lp_t;
  gp_t gp = (gp_t)(size_t)g;
  unsigned loff = (unsigned)(size_t)l;  // low 32 bits of generic shared addr = LDS offset
  lp_t lp = (lp_t)loff;
  __builtin_amdgcn_global_load_async_to_lds_b128(gp, lp, 0, 0);
#else
  *(v8bf*)l = *(const v8bf*)g;
#endif
}
static __device__ inline void stage_wait() {
#if HAVE_ASYNC_LDS
  __builtin_amdgcn_s_wait_asynccnt(0);
#endif
}

static __device__ inline __bf16 f2bf(float f) {
  unsigned u = __builtin_bit_cast(unsigned, f);
  unsigned r = (u + 0x7FFFu + ((u >> 16) & 1u)) >> 16;
  unsigned short s = (unsigned short)r;
  return __builtin_bit_cast(__bf16, s);
}

// fast transcendentals -> v_exp_f32 / v_log_f32 TRANS ops
static __device__ inline float fexp(float x) { return __expf(x); }
static __device__ inline float fsoftplus(float x) {
  return (x > 20.0f) ? x : __logf(1.0f + __expf(x));
}

// ---- packed-bf16 WMMA fragment loaders (16x16x32) ----
// A 16x32 (ISA 7.12.2): lane l -> row m=l&15, half h=l>>4.
//   elems 0..15 : K = 16h+e (contiguous)
static __device__ inline v16bf load_a_bf(const __bf16* p, int ld, int r0, int k0,
                                         int lane) {
  int m = lane & 15, h = lane >> 4;
  const __bf16* base = p + (size_t)(r0 + m) * ld + k0 + 16 * h;
  v8bf lo = *(const v8bf*)(base);
  v8bf hi = *(const v8bf*)(base + 8);
  return __builtin_shufflevector(lo, hi, 0, 1, 2, 3, 4, 5, 6, 7, 8, 9, 10, 11, 12,
                                 13, 14, 15);
}
// B 32x16: lane l -> col n=l&15, half h=l>>4 ; elems 0..15 : K = 16h+e (contiguous)
static __device__ inline v16bf load_b_bf(const __bf16* p, int ld, int k0, int c0,
                                         int lane) {
  int n = lane & 15, h = lane >> 4;
  const __bf16* base = p + (size_t)(c0 + n) * ld + k0 + 16 * h;
  v8bf lo = *(const v8bf*)(base);
  v8bf hi = *(const v8bf*)(base + 8);
  return __builtin_shufflevector(lo, hi, 0, 1, 2, 3, 4, 5, 6, 7, 8, 9, 10, 11, 12,
                                 13, 14, 15);
}
// B fragment from an LDS-staged tile laid out [64 cols][BKP K-elems]
static __device__ inline v16bf load_b_lds(const __bf16* Bs, int c, int lane) {
  int n = lane & 15, h = lane >> 4;
  const __bf16* base = Bs + (c * 16 + n) * BKP + 16 * h;
  v8bf lo = *(const v8bf*)(base);
  v8bf hi = *(const v8bf*)(base + 8);
  return __builtin_shufflevector(lo, hi, 0, 1, 2, 3, 4, 5, 6, 7, 8, 9, 10, 11, 12,
                                 13, 14, 15);
}

static __device__ inline v8f wmma_bf16(v16bf a, v16bf b, v8f c) {
  return __builtin_amdgcn_wmma_f32_16x16x32_bf16(false, a, false, b, (short)0, c,
                                                 false, false);
}

// ---------------- K1: per-module precompute + bf16 packing ----------------
__global__ void k_prep(const float* __restrict__ z, const float* __restrict__ z_j,
                       const float* __restrict__ vec_d, const float* __restrict__ spr,
                       const float* __restrict__ sppr, float* __restrict__ wperp,
                       float* __restrict__ wdiff, float* __restrict__ zjzj,
                       float* __restrict__ zbv, float* __restrict__ zz,
                       float* __restrict__ accum, __bf16* __restrict__ z_bf,
                       __bf16* __restrict__ zj_bf, __bf16* __restrict__ bvec_bf) {
  int m = blockIdx.x * blockDim.x + threadIdx.x;
  if (m == 0) accum[0] = 0.0f;
  if (m >= M_DIM) return;
  const float EPSF = 1.1920929e-7f;  // float32 eps
  float sp  = fmaxf(fsoftplus(spr[m])  + EPSF, EPSF);
  float spp = fmaxf(fsoftplus(sppr[m]) + EPSF, EPSF);
  float wpar = 1.0f / (sp * sp);
  float wper = 1.0f / (spp * spp);
  wperp[m] = wper;
  wdiff[m] = wpar - wper;

  float dn2 = 0.0f;
  for (int n = 0; n < N_DIM; ++n) { float v = vec_d[m * N_DIM + n]; dn2 += v * v; }
  float dn = sqrtf(dn2);
  float inv = (dn > 1e-6f) ? (1.0f / dn) : 0.0f;

  float zj2 = 0.0f, zb = 0.0f;
  for (int n = 0; n < N_DIM; ++n) {
    float bv = vec_d[m * N_DIM + n] * inv;
    float zv = z_j[m * N_DIM + n];
    bvec_bf[m * N_DIM + n] = f2bf(bv);
    zj_bf[m * N_DIM + n] = f2bf(zv);
    zj2 += zv * zv;
    zb += zv * bv;
  }
  zjzj[m] = zj2;
  zbv[m] = zb;

  if (m < B_DIM) {
    float s = 0.0f;
    for (int n = 0; n < N_DIM; ++n) {
      float v = z[m * N_DIM + n];
      z_bf[m * N_DIM + n] = f2bf(v);
      s += v * v;
    }
    zz[m] = s;
  }
}

// ---------------- K1b: pack T_hat^T as bf16 [S][M] (coalesced b16 stores) ---------
__global__ void k_tpose(const float* __restrict__ T_hat, __bf16* __restrict__ tht_t) {
  int i = blockIdx.x * blockDim.x + threadIdx.x;  // i = s*M + m, m fastest
  if (i >= M_DIM * S_DIM) return;
  int m = i & (M_DIM - 1), s = i >> 12;
  tht_t[i] = f2bf(T_hat[m * S_DIM + s]);
}

// ------- K2: gain via K=64 WMMA products, 16(B)x32(M) strip per wave -------
__global__ void k_gain(const __bf16* __restrict__ z_bf, const __bf16* __restrict__ zj_bf,
                       const __bf16* __restrict__ bvec_bf,
                       const float* __restrict__ wperp, const float* __restrict__ wdiff,
                       const float* __restrict__ zjzj, const float* __restrict__ zbv,
                       const float* __restrict__ zz, const float* __restrict__ alpha_j,
                       __bf16* __restrict__ gain_bf, __bf16* __restrict__ gain_t_bf) {
  int lane = threadIdx.x & 31;
  int t = blockIdx.x * 8 + (threadIdx.x >> 5);  // 4096 waves = 32 (B) x 128 (M/32)
  int tb = t >> 7, tm = t & 127;
  int r0 = tb * 16, c0 = tm * 32;

  v8f accZ[2] = {{}, {}}, accP[2] = {{}, {}};
#pragma unroll
  for (int k0 = 0; k0 < N_DIM; k0 += 32) {
    v16bf a = load_a_bf(z_bf, N_DIM, r0, k0, lane);
#pragma unroll
    for (int c = 0; c < 2; ++c) {
      v16bf bz = load_b_bf(zj_bf, N_DIM, k0, c0 + 16 * c, lane);
      v16bf bb = load_b_bf(bvec_bf, N_DIM, k0, c0 + 16 * c, lane);
      accZ[c] = wmma_bf16(a, bz, accZ[c]);
      accP[c] = wmma_bf16(a, bb, accP[c]);
    }
  }
  int h = lane >> 4;
#pragma unroll
  for (int c = 0; c < 2; ++c) {
    int mcol = c0 + 16 * c + (lane & 15);
    float wp = wperp[mcol], wd = wdiff[mcol], zj2 = zjzj[mcol], zb = zbv[mcol];
    float al = alpha_j[mcol];
#pragma unroll
    for (int r = 0; r < 8; ++r) {
      int brow = r0 + r + 8 * h;
      float dzn = zz[brow] + zj2 - 2.0f * accZ[c][r];
      float pr = accP[c][r] - zb;
      float q = wp * dzn + wd * pr * pr;
      float qpos = 25.0f - fsoftplus(25.0f - q);  // smooth clamp at MAX_Q
      float g = al * fexp(-3.14159265358979f * qpos);
      __bf16 gb = f2bf(g);
      gain_bf[brow * M_DIM + mcol] = gb;
      gain_t_bf[mcol * B_DIM + brow] = gb;
    }
  }
}

// ------- K3: E^T = (gain @ T_hat - T_star)^T bf16 [S][B] -------
// Block = 8 waves covering 128 B-rows x 64 S-cols; B tile staged in LDS
// (double-buffered, one 16B global->LDS copy per thread per step — async DMA
// where available) and shared by all 8 waves. Prefetch 2 steps ahead.
__global__ void k_err(const __bf16* __restrict__ gain_bf,
                      const __bf16* __restrict__ tht_t, const float* __restrict__ T_star,
                      __bf16* __restrict__ et_bf) {
  __shared__ __bf16 Bs[2][64 * BKP];
  int tid = threadIdx.x;
  int lane = tid & 31, wave = tid >> 5;
  int rb = blockIdx.x >> 2, ts = blockIdx.x & 3;  // 16 blocks = 4 (B/128) x 4 (S/64)
  int r0 = rb * 128 + wave * 16;
  int c0 = ts * 64;

  int lcol = tid >> 2, lk = (tid & 3) * 8;  // 64 cols x 4 chunks of 8 bf16
  const __bf16* gsrc = tht_t + (size_t)(c0 + lcol) * M_DIM + lk;
  int ldst = lcol * BKP + lk;

  stage_b128(gsrc, &Bs[0][ldst]);
  stage_wait();
  __syncthreads();

  v8f acc[4] = {{}, {}, {}, {}};
  const int NSTEP = M_DIM / BK;  // 128
  for (int step = 0; step < NSTEP; ++step) {
    int buf = step & 1;
    if (step + 1 < NSTEP)
      stage_b128(gsrc + (size_t)(step + 1) * BK, &Bs[buf ^ 1][ldst]);
    if (step + 2 < NSTEP)
      __builtin_prefetch(gsrc + (size_t)(step + 2) * BK, 0, 0);
    int k0 = step * BK;
    v16bf a = load_a_bf(gain_bf, M_DIM, r0, k0, lane);
#pragma unroll
    for (int c = 0; c < 4; ++c) {
      v16bf b = load_b_lds(&Bs[buf][0], c, lane);
      acc[c] = wmma_bf16(a, b, acc[c]);
    }
    stage_wait();
    __syncthreads();
  }
  int h = lane >> 4;
#pragma unroll
  for (int c = 0; c < 4; ++c) {
    int scol = c0 + 16 * c + (lane & 15);
#pragma unroll
    for (int r = 0; r < 8; ++r) {
      int brow = r0 + r + 8 * h;
      et_bf[scol * B_DIM + brow] = f2bf(acc[c][r] - T_star[brow * S_DIM + scol]);
    }
  }
}

// ------- K4: delta^T = -sigmoid(alog)*(gain^T @ E)^T/B ; accumulate ||delta||^2 ----
__global__ void k_grad(const __bf16* __restrict__ gain_t_bf,
                       const __bf16* __restrict__ et_bf, const float* __restrict__ alog,
                       float* __restrict__ delta_t, float* __restrict__ accum) {
  int lane = threadIdx.x & 31;
  int t = blockIdx.x * 8 + (threadIdx.x >> 5);  // 1024 waves = 256 (M) x 4 (S/64)
  int tm = t >> 2, ts = t & 3;
  int r0 = tm * 16, c0 = ts * 64;
  v8f acc[4] = {{}, {}, {}, {}};
#pragma unroll 2
  for (int k0 = 0; k0 < B_DIM; k0 += 32) {
    v16bf a = load_a_bf(gain_t_bf, B_DIM, r0, k0, lane);
#pragma unroll
    for (int c = 0; c < 4; ++c) {
      v16bf b = load_b_bf(et_bf, B_DIM, k0, c0 + 16 * c, lane);
      acc[c] = wmma_bf16(a, b, acc[c]);
    }
  }
  float asig = 1.0f / (1.0f + fexp(-alog[0]));
  float scale = -asig * (1.0f / (float)B_DIM);
  int h = lane >> 4;
  float lsum = 0.0f;
#pragma unroll
  for (int c = 0; c < 4; ++c) {
    int scol = c0 + 16 * c + (lane & 15);
#pragma unroll
    for (int r = 0; r < 8; ++r) {
      int mrow = r0 + r + 8 * h;
      float d = scale * acc[c][r];
      delta_t[scol * M_DIM + mrow] = d;  // stored transposed [S][M]
      lsum += d * d;
    }
  }
  for (int off = 16; off > 0; off >>= 1) lsum += __shfl_xor(lsum, off, 32);
  if (lane == 0) atomicAdd(accum, lsum);
}

// ------- K5: Frobenius cap; thd^T = T_hat^T + delta^T * s  (pure linear pass) ------
__global__ void k_scale(const __bf16* __restrict__ tht_t, const float* __restrict__ accum,
                        const float* __restrict__ delta_t, __bf16* __restrict__ thd_t) {
  int i = blockIdx.x * blockDim.x + threadIdx.x;
  if (i >= M_DIM * S_DIM) return;
  const float TINY = 1.17549435e-38f;
  float nrm = sqrtf(accum[0]);
  float s = fminf(1.0f / (nrm + TINY), 1.0f);  // CAP = 1.0
  thd_t[i] = f2bf((float)tht_t[i] + delta_t[i] * s);
}

// ---------------- K6: T = gain @ (T_hat + delta); LDS-staged like K3 ---------------
__global__ void k_final(const __bf16* __restrict__ gain_bf,
                        const __bf16* __restrict__ thd_t, float* __restrict__ out) {
  __shared__ __bf16 Bs[2][64 * BKP];
  int tid = threadIdx.x;
  int lane = tid & 31, wave = tid >> 5;
  int rb = blockIdx.x >> 2, ts = blockIdx.x & 3;
  int r0 = rb * 128 + wave * 16;
  int c0 = ts * 64;

  int lcol = tid >> 2, lk = (tid & 3) * 8;
  const __bf16* gsrc = thd_t + (size_t)(c0 + lcol) * M_DIM + lk;
  int ldst = lcol * BKP + lk;

  stage_b128(gsrc, &Bs[0][ldst]);
  stage_wait();
  __syncthreads();

  v8f acc[4] = {{}, {}, {}, {}};
  const int NSTEP = M_DIM / BK;  // 128
  for (int step = 0; step < NSTEP; ++step) {
    int buf = step & 1;
    if (step + 1 < NSTEP)
      stage_b128(gsrc + (size_t)(step + 1) * BK, &Bs[buf ^ 1][ldst]);
    if (step + 2 < NSTEP)
      __builtin_prefetch(gsrc + (size_t)(step + 2) * BK, 0, 0);
    int k0 = step * BK;
    v16bf a = load_a_bf(gain_bf, M_DIM, r0, k0, lane);
#pragma unroll
    for (int c = 0; c < 4; ++c) {
      v16bf b = load_b_lds(&Bs[buf][0], c, lane);
      acc[c] = wmma_bf16(a, b, acc[c]);
    }
    stage_wait();
    __syncthreads();
  }
  int h = lane >> 4;
#pragma unroll
  for (int c = 0; c < 4; ++c) {
    int scol = c0 + 16 * c + (lane & 15);
#pragma unroll
    for (int r = 0; r < 8; ++r) {
      int brow = r0 + r + 8 * h;
      out[brow * S_DIM + scol] = acc[c][r];
    }
  }
}

extern "C" void kernel_launch(void* const* d_in, const int* in_sizes, int n_in,
                              void* d_out, int out_size, void* d_ws, size_t ws_size,
                              hipStream_t stream) {
  const float* z       = (const float*)d_in[0];
  const float* T_star  = (const float*)d_in[1];
  const float* z_j     = (const float*)d_in[2];
  const float* vec_d_j = (const float*)d_in[3];
  const float* T_hat_j = (const float*)d_in[4];
  const float* alpha_j = (const float*)d_in[5];
  const float* spr     = (const float*)d_in[6];
  const float* sppr    = (const float*)d_in[7];
  const float* alog    = (const float*)d_in[8];
  float* out = (float*)d_out;

  // ---- workspace layout: f32 region first, then bf16 region ----
  float* ws = (float*)d_ws;
  float* wperp   = ws;  ws += M_DIM;
  float* wdiff   = ws;  ws += M_DIM;
  float* zjzj    = ws;  ws += M_DIM;
  float* zbv     = ws;  ws += M_DIM;
  float* zz      = ws;  ws += B_DIM;
  float* accum   = ws;  ws += 8;              // padded for alignment
  float* delta_t = ws;  ws += M_DIM * S_DIM;  // 1048576 f32, stored [S][M]

  __bf16* bw = (__bf16*)ws;
  __bf16* z_bf      = bw;  bw += B_DIM * N_DIM;   //   32768
  __bf16* zj_bf     = bw;  bw += M_DIM * N_DIM;   //  262144
  __bf16* bvec_bf   = bw;  bw += M_DIM * N_DIM;   //  262144
  __bf16* gain_bf   = bw;  bw += B_DIM * M_DIM;   // 2097152
  __bf16* gain_t_bf = bw;  bw += M_DIM * B_DIM;   // 2097152
  __bf16* tht_t_bf  = bw;  bw += S_DIM * M_DIM;   // 1048576
  __bf16* et_bf     = bw;  bw += S_DIM * B_DIM;   //  131072
  __bf16* thd_t_bf  = bw;  bw += S_DIM * M_DIM;   // 1048576

  k_prep<<<M_DIM / 256, 256, 0, stream>>>(z, z_j, vec_d_j, spr, sppr, wperp, wdiff,
                                          zjzj, zbv, zz, accum, z_bf, zj_bf, bvec_bf);
  k_tpose<<<(M_DIM * S_DIM) / 256, 256, 0, stream>>>(T_hat_j, tht_t_bf);
  // 4096 waves / 8 per block
  k_gain<<<512, 256, 0, stream>>>(z_bf, zj_bf, bvec_bf, wperp, wdiff, zjzj, zbv, zz,
                                  alpha_j, gain_bf, gain_t_bf);
  // 16 blocks x 8 waves (LDS-staged)
  k_err<<<16, 256, 0, stream>>>(gain_bf, tht_t_bf, T_star, et_bf);
  // 1024 waves
  k_grad<<<128, 256, 0, stream>>>(gain_t_bf, et_bf, alog, delta_t, accum);
  k_scale<<<(M_DIM * S_DIM) / 256, 256, 0, stream>>>(tht_t_bf, accum, delta_t,
                                                     thd_t_bf);
  // 16 blocks x 8 waves (LDS-staged)
  k_final<<<16, 256, 0, stream>>>(gain_bf, thd_t_bf, out);
}
